// RNNImplemented_55241869361399
// MI455X (gfx1250) — compile-verified, matrix operand order checked by reading, not run on hardware
//
#include <hip/hip_runtime.h>
#include <math.h>

// ---------------------------------------------------------------------------
// RNN forward on MI455X (gfx1250), wave32 + v_wmma_f32_16x16x32_bf16.
//   xi = x @ W_ih^T + b_ih          (big GEMM, written into d_out in place)
//   h_t = tanh(xi_t + h_{t-1} @ W_hh^T + b_hh)   (persistent cooperative scan,
//   W_hh tiles cached in LDS, one 16x16 tile per wave, grid barrier per step)
// ---------------------------------------------------------------------------

#define B_  64
#define T_  512
#define C_  512
#define H_  1024
#define BT_ (B_ * T_)

// LDS W-tile row stride (bf16 elems): 1024 + 8 pad -> 2064 B; 2064 % 256 == 16
// so the 16 lanes of a fragment read cover all 64 LDS banks conflict-free.
#define WROW_ 1032
#define LDS_PER_WAVE_ (16 * WROW_)               // bf16 elems
#define LDS_BYTES_ (8 * LDS_PER_WAVE_ * 2)       // 8 waves/block -> 264,192 B

#if __has_builtin(__builtin_amdgcn_global_load_async_to_lds_b128) && \
    __has_builtin(__builtin_amdgcn_s_wait_asynccnt)
#define HAVE_ASYNC_LDS_ 1
#else
#define HAVE_ASYNC_LDS_ 0
#endif

typedef __attribute__((ext_vector_type(16))) __bf16 v16bf;
typedef __attribute__((ext_vector_type(4)))  __bf16 v4bf;
typedef __attribute__((ext_vector_type(8)))  float  v8f;
typedef int v4i_ __attribute__((vector_size(16)));
typedef __attribute__((address_space(1))) v4i_* gv4p_;   // global int4*
typedef __attribute__((address_space(3))) v4i_* lv4p_;   // LDS int4*

// Workspace layout (bytes)
#define OFF_XB   ((size_t)0)                          // bf16 x     [BT, C]   33.5 MB
#define OFF_WIH  (OFF_XB  + (size_t)2 * BT_ * C_)     // bf16 W_ih  [H, C]     1 MB
#define OFF_WHH  (OFF_WIH + (size_t)2 * H_ * C_)      // bf16 W_hh  [H, H]     2 MB
#define OFF_HB   (OFF_WHH + (size_t)2 * H_ * H_)      // bf16 h     [2, B, H]  256 KB
#define OFF_CTR  (OFF_HB  + (size_t)2 * 2 * B_ * H_)  // u32 barrier ctr [T]   2 KB

// ---------------------------------------------------------------------------
// Fragment loader for the 16-bit A/B operand of V_WMMA_*_16X16X32_BF16.
// ISA 7.12.2: lane L holds row (L&15); VGPRs 0..3 carry K = kb + 8*(L>>4)+0..7,
// VGPRs 4..7 carry K = kb + 16 + 8*(L>>4)+0..7 -> two contiguous b128 loads.
// ---------------------------------------------------------------------------
__device__ __forceinline__ v16bf load_frag16(const __bf16* rowptr,
                                             int kbase, int laneHi) {
  union { v16bf v; uint4 q[2]; } f;
  const __bf16* p = rowptr + kbase + laneHi * 8;
  f.q[0] = *reinterpret_cast<const uint4*>(p);       // *_load_b128
  f.q[1] = *reinterpret_cast<const uint4*>(p + 16);  // *_load_b128
  return f.v;
}

__device__ __forceinline__ v8f wmma_bf16(v16bf a, v16bf b, v8f c) {
  return __builtin_amdgcn_wmma_f32_16x16x32_bf16(
      /*neg_a=*/false, a, /*neg_b=*/false, b,
      /*c_mod=*/(short)0, c, /*reuse_a=*/false, /*reuse_b=*/false);
}

// ---------------------------------------------------------------------------
// Prep: fp32 -> bf16 conversion (4-wide), h0 broadcast + barrier counter init
// ---------------------------------------------------------------------------
__global__ void cvt_bf16_v4(const float* __restrict__ s, __bf16* __restrict__ d,
                            size_t n4) {
  size_t i  = (size_t)blockIdx.x * blockDim.x + threadIdx.x;
  size_t st = (size_t)gridDim.x * blockDim.x;
  const float4* s4 = reinterpret_cast<const float4*>(s);
  for (; i < n4; i += st) {
    float4 f = s4[i];
    v4bf o = { (__bf16)f.x, (__bf16)f.y, (__bf16)f.z, (__bf16)f.w };
    *reinterpret_cast<v4bf*>(d + 4 * i) = o;
  }
}

__global__ void init_state(const float* __restrict__ h0,
                           __bf16* __restrict__ hbuf,
                           unsigned* __restrict__ ctr) {
  int i = blockIdx.x * blockDim.x + threadIdx.x;
  if (i < B_ * H_) hbuf[i] = (__bf16)h0[i & (H_ - 1)];  // broadcast over batch
  if (i < T_)      ctr[i] = 0u;
}

// ---------------------------------------------------------------------------
// Phase 1: xi = x @ W_ih^T + b_ih -> d_out[b,t,:] (overwritten later by h_t).
// One wave per block; 32 rows x 64 cols per wave (8 wmma tiles): each B
// fragment feeds two WMMAs, each A fragment feeds four.
// ---------------------------------------------------------------------------
__global__ void __launch_bounds__(32)
gemm_input(const __bf16* __restrict__ xb, const __bf16* __restrict__ wih,
           const float* __restrict__ bih, float* __restrict__ out) {
  const int lane = threadIdx.x;
  const int lm = lane & 15, hf = lane >> 4;
  const int r0 = blockIdx.x * 32;   // row tile in [BT]
  const int n0 = blockIdx.y * 64;   // col group in [H]

  const __bf16* arow0 = xb + (size_t)(r0 + lm) * C_;
  const __bf16* arow1 = xb + (size_t)(r0 + 16 + lm) * C_;
  v8f acc[2][4] = {};

  for (int k = 0; k < C_; k += 32) {
    v16bf a0 = load_frag16(arow0, k, hf);
    v16bf a1 = load_frag16(arow1, k, hf);
#pragma unroll
    for (int j = 0; j < 4; ++j) {
      v16bf b = load_frag16(wih + (size_t)(n0 + 16 * j + lm) * C_, k, hf);
      acc[0][j] = wmma_bf16(a0, b, acc[0][j]);
      acc[1][j] = wmma_bf16(a1, b, acc[1][j]);
    }
  }

  // C/D layout: lane L -> col (L&15); VGPR i -> row i + 8*(L>>4)
#pragma unroll
  for (int j = 0; j < 4; ++j) {
    const int n = n0 + 16 * j + lm;
    const float bv = bih[n];
#pragma unroll
    for (int i = 0; i < 8; ++i) {
      const int r = r0 + 8 * hf + i;
      out[(size_t)r * H_ + n]        = acc[0][j][i] + bv;
      out[(size_t)(r + 16) * H_ + n] = acc[1][j][i] + bv;
    }
  }
}

// ---------------------------------------------------------------------------
// Phase 2: persistent scan. 32 blocks x 8 waves = 256 waves; one 16x16 tile
// of h [B=64, H=1024] per wave. W_hh slice staged to LDS once (async global->
// LDS when available), B fragments via ds_load_b128, A (h_prev) fragments
// from global with one-deep register prefetch, xi preloaded ahead of the K
// loop so its latency hides under the WMMA stream. Grid barrier per step.
// ---------------------------------------------------------------------------
__global__ void __launch_bounds__(256)
rnn_scan(const __bf16* __restrict__ whh, const float* __restrict__ bhh,
         float* __restrict__ out,          // [B,T,H]: xi in, h_t out (in place)
         __bf16* __restrict__ hbuf,        // [2,B,H] bf16 ping-pong
         unsigned* __restrict__ ctr,       // [T] arrival counters (pre-zeroed)
         float* __restrict__ hlast) {      // [B,H]
  extern __shared__ __bf16 smem[];         // 8 * 16 * WROW_ bf16

  const int lane = threadIdx.x & 31;
  const int wid  = threadIdx.x >> 5;
  const int wg   = blockIdx.x * 8 + wid;   // 0..255
  const int mi   = wg >> 6;                // 0..3   batch tile
  const int ni   = wg & 63;                // 0..63  hidden tile
  const int lm = lane & 15, hf = lane >> 4;

  const int arowi = mi * 16 + lm;          // h row (batch index) for A frag
  const int n     = ni * 16 + lm;          // output column (hidden index)
  const float bias = bhh[n];
  const unsigned nblk = gridDim.x;

  // ---- one-time stage of this wave's W_hh slice into LDS ----------------
  __bf16* wl = smem + (size_t)wid * LDS_PER_WAVE_;
  {
    // lane lm handles row lm; halves hf cover k in [0,512) / [512,1024)
    const __bf16* g = whh + (size_t)(ni * 16 + lm) * H_ + hf * 512;
    __bf16* l = wl + (size_t)lm * WROW_ + hf * 512;
#if HAVE_ASYNC_LDS_
#pragma unroll 4
    for (int c = 0; c < 512; c += 8)
      __builtin_amdgcn_global_load_async_to_lds_b128(
          (gv4p_)(g + c), (lv4p_)(l + c), 0, 0);  // global_load_async_to_lds
    __builtin_amdgcn_s_wait_asynccnt(0);          // s_wait_asynccnt 0
#else
#pragma unroll 4
    for (int c = 0; c < 512; c += 8)
      *reinterpret_cast<uint4*>(l + c) =
          *reinterpret_cast<const uint4*>(g + c);   // global_load + ds_store
#endif
  }
  const __bf16* wrowL = wl + (size_t)lm * WROW_;    // B frag base (row lm)

  for (int t = 0; t < T_; ++t) {
    const __bf16* hp = hbuf + (size_t)(t & 1) * B_ * H_;
    __bf16*       hn = hbuf + (size_t)((t + 1) & 1) * B_ * H_;
    const __bf16* arow = hp + (size_t)arowi * H_;

    // Preload xi for this step's tile: independent of the GEMM, so these 8
    // strided loads retire underneath the 32-WMMA K loop.
    size_t oidx[8];
    float  xi[8];
#pragma unroll
    for (int i = 0; i < 8; ++i) {
      const int row = mi * 16 + 8 * hf + i;                 // batch index
      oidx[i] = ((size_t)row * T_ + t) * H_ + n;            // hs[row, t, n]
      xi[i] = out[oidx[i]];
    }

    v8f acc = {};
    v16bf a = load_frag16(arow, 0, hf);             // software pipeline: A
    for (int k = 0; k < H_; k += 32) {
      v16bf a_next;
      if (k + 32 < H_) {
        __builtin_prefetch(arow + k + 64, 0, 1);    // global_prefetch_b8
        a_next = load_frag16(arow, k + 32, hf);
      }
      // B fragment from LDS: row lm, same K swizzle as global layout
      union { v16bf v; uint4 q[2]; } bf;
      const __bf16* p = wrowL + k + hf * 8;
      bf.q[0] = *reinterpret_cast<const uint4*>(p);       // ds_load_b128
      bf.q[1] = *reinterpret_cast<const uint4*>(p + 16);  // ds_load_b128
      acc = wmma_bf16(a, bf.v, acc);
      a = a_next;
    }

#pragma unroll
    for (int i = 0; i < 8; ++i) {
      const int row = mi * 16 + 8 * hf + i;                 // batch index
      const float v = tanhf(acc[i] + xi[i] + bias);
      out[oidx[i]] = v;
      hn[(size_t)row * H_ + n] = (__bf16)v;
      if (t == T_ - 1) hlast[(size_t)row * H_ + n] = v;
    }

    // grid-wide barrier for step t
    __threadfence();
    __syncthreads();
    if (threadIdx.x == 0) {
      __hip_atomic_fetch_add(&ctr[t], 1u, __ATOMIC_RELEASE,
                             __HIP_MEMORY_SCOPE_AGENT);
      while (__hip_atomic_load(&ctr[t], __ATOMIC_ACQUIRE,
                               __HIP_MEMORY_SCOPE_AGENT) < nblk) { }
    }
    __syncthreads();
  }
}

// ---------------------------------------------------------------------------
extern "C" void kernel_launch(void* const* d_in, const int* in_sizes, int n_in,
                              void* d_out, int out_size, void* d_ws,
                              size_t ws_size, hipStream_t stream) {
  const float* x   = (const float*)d_in[0];  // [B,T,C]
  const float* Wih = (const float*)d_in[1];  // [H,C]
  const float* bih = (const float*)d_in[2];  // [H]
  const float* Whh = (const float*)d_in[3];  // [H,H]
  const float* bhh = (const float*)d_in[4];  // [H]
  const float* h0  = (const float*)d_in[5];  // [H]

  char* ws = (char*)d_ws;
  __bf16*   xb   = (__bf16*)(ws + OFF_XB);
  __bf16*   wihb = (__bf16*)(ws + OFF_WIH);
  __bf16*   whhb = (__bf16*)(ws + OFF_WHH);
  __bf16*   hbuf = (__bf16*)(ws + OFF_HB);
  unsigned* ctr  = (unsigned*)(ws + OFF_CTR);

  float* out   = (float*)d_out;                 // hs [B,T,H]
  float* hlast = out + (size_t)B_ * T_ * H_;    // h_last [B,H]

  // 1) precision prep + state init
  cvt_bf16_v4<<<4096, 256, 0, stream>>>(x,   xb,   (size_t)BT_ * C_ / 4);
  cvt_bf16_v4<<<512,  256, 0, stream>>>(Wih, wihb, (size_t)H_ * C_ / 4);
  cvt_bf16_v4<<<512,  256, 0, stream>>>(Whh, whhb, (size_t)H_ * H_ / 4);
  init_state<<<(B_ * H_ + 255) / 256, 256, 0, stream>>>(h0, hbuf, ctr);

  // 2) input projection into d_out (xi occupies hs[:,t,:] until step t)
  dim3 g1(BT_ / 32, H_ / 64);
  gemm_input<<<g1, 32, 0, stream>>>(xb, wihb, bih, out);

  // 3) persistent cooperative recurrent scan (W_hh cached in 258 KB of LDS)
  rnn_scan<<<32, 256, LDS_BYTES_, stream>>>(whhb, bhh, out, hbuf, ctr, hlast);
}
